// SeparateProjectionLayer_41661182771240
// MI455X (gfx1250) — matrix operand order repaired and checked by reading, not compiled, not run on hardware
//
#include <hip/hip_runtime.h>

typedef __attribute__((ext_vector_type(16))) __bf16 v16bf;
typedef __attribute__((ext_vector_type(8)))  __bf16 v8bf;
typedef __attribute__((ext_vector_type(8)))  float  v8f;

namespace {
constexpr int SEQ   = 512;
constexpr int BATCH = 128;
constexpr int FEAT  = 1024;
constexpr int EMBED = 1024;

constexpr int BM  = 128;            // M tile (seq)
constexpr int BN  = 128;            // N tile (embed)
constexpr int BK  = 32;             // K tile (one WMMA K-step)
constexpr int NKT = FEAT / BK;      // 32 K iterations
constexpr int LDT   = BK + 8;       // padded LDS row stride in halves (80B: 16B-aligned, conflict-free)
constexpr int PLANE = BM * LDT;     // halves per tile plane (5120)
constexpr int STAGE = 4 * PLANE;    // A_hi, A_lo, B_hi, B_lo
}

__global__ __launch_bounds__(256)
void grouped_linear_bf16x3_wmma(const float* __restrict__ feat,
                                const float* __restrict__ Wt,
                                const float* __restrict__ bias,
                                const int*  __restrict__ toks,
                                float* __restrict__ out)
{
    __shared__ __bf16 smem[2 * STAGE];   // 80 KB double-buffered

    const int b     = blockIdx.z;
    const int t     = toks[b];
    const int sBase = blockIdx.y * BM;
    const int eBase = blockIdx.x * BN;

    const int tid  = threadIdx.x;
    const int lane = tid & 31;
    const int wid  = tid >> 5;
    const int lh   = lane >> 4;     // lane-half selector
    const int ln   = lane & 15;     // row-in-tile / n-in-tile

    // 8 waves: 4 along M (32 rows each) x 2 along N (64 cols each)
    const int wm = (wid & 3) * 32;
    const int wn = (wid >> 2) * 64;

    // ---- staging map: each thread loads 16 consecutive floats of one row ----
    const int lrow = tid >> 1;           // 0..127
    const int lcol = (tid & 1) * 16;     // 0 or 16

    const float* gA = feat + ((size_t)(sBase + lrow) * BATCH + b) * FEAT + lcol;
    const float* gB = Wt   + ((size_t)t * EMBED + (eBase + lrow)) * FEAT + lcol;
    const int ldsOff = lrow * LDT + lcol;

    v8f acc[2][4];
#pragma unroll
    for (int i = 0; i < 2; ++i)
#pragma unroll
        for (int j = 0; j < 4; ++j)
            acc[i][j] = (v8f){0.f,0.f,0.f,0.f,0.f,0.f,0.f,0.f};

    float4 ra[4], rb[4];

    auto gload = [&](int kt) {
        const float4* pa = reinterpret_cast<const float4*>(gA + kt * BK);
        const float4* pb = reinterpret_cast<const float4*>(gB + kt * BK);
#pragma unroll
        for (int i = 0; i < 4; ++i) { ra[i] = pa[i]; rb[i] = pb[i]; }
    };

    auto stage_store = [&](int st) {
        __bf16* aHi = smem + st * STAGE;
        __bf16* aLo = aHi + PLANE;
        __bf16* bHi = aLo + PLANE;
        __bf16* bLo = bHi + PLANE;
        float fa[16], fb[16];
#pragma unroll
        for (int i = 0; i < 4; ++i) {
            fa[4*i+0] = ra[i].x; fa[4*i+1] = ra[i].y; fa[4*i+2] = ra[i].z; fa[4*i+3] = ra[i].w;
            fb[4*i+0] = rb[i].x; fb[4*i+1] = rb[i].y; fb[4*i+2] = rb[i].z; fb[4*i+3] = rb[i].w;
        }
#pragma unroll
        for (int h = 0; h < 2; ++h) {
            v8bf ah, al, bh, bl;
#pragma unroll
            for (int j = 0; j < 8; ++j) {
                float xa = fa[8*h + j];
                __bf16 ha = (__bf16)xa;
                ah[j] = ha;
                al[j] = (__bf16)(xa - (float)ha);
                float xb = fb[8*h + j];
                __bf16 hb = (__bf16)xb;
                bh[j] = hb;
                bl[j] = (__bf16)(xb - (float)hb);
            }
            *reinterpret_cast<v8bf*>(aHi + ldsOff + 8*h) = ah;
            *reinterpret_cast<v8bf*>(aLo + ldsOff + 8*h) = al;
            *reinterpret_cast<v8bf*>(bHi + ldsOff + 8*h) = bh;
            *reinterpret_cast<v8bf*>(bLo + ldsOff + 8*h) = bl;
        }
    };

    auto compute = [&](int st) {
        const __bf16* aHi = smem + st * STAGE;
        const __bf16* aLo = aHi + PLANE;
        const __bf16* bHi = aLo + PLANE;
        const __bf16* bLo = bHi + PLANE;

        v16bf Ah[2], Al[2], Bh[4], Bl[4];
#pragma unroll
        for (int mt = 0; mt < 2; ++mt) {
            // A 16x32 bf16 layout: lane L: K=lh*8..+7 and K=16+lh*8..+7
            const int row = wm + mt * 16 + ln;
            const int o0 = row * LDT + lh * 8;
            const int o1 = row * LDT + 16 + lh * 8;
            v8bf p0 = *reinterpret_cast<const v8bf*>(aHi + o0);
            v8bf p1 = *reinterpret_cast<const v8bf*>(aHi + o1);
            Ah[mt] = __builtin_shufflevector(p0, p1, 0,1,2,3,4,5,6,7,8,9,10,11,12,13,14,15);
            p0 = *reinterpret_cast<const v8bf*>(aLo + o0);
            p1 = *reinterpret_cast<const v8bf*>(aLo + o1);
            Al[mt] = __builtin_shufflevector(p0, p1, 0,1,2,3,4,5,6,7,8,9,10,11,12,13,14,15);
        }
#pragma unroll
        for (int nt = 0; nt < 4; ++nt) {
            // B 32x16 bf16 layout: lane L: K=lh*8..+7 and K=16+lh*8..+7 (per column)
            const int row = wn + nt * 16 + ln;            // B stored N x K
            const int o0 = row * LDT + lh * 8;
            v8bf p0 = *reinterpret_cast<const v8bf*>(bHi + o0);
            v8bf p1 = *reinterpret_cast<const v8bf*>(bHi + o0 + 16);
            Bh[nt] = __builtin_shufflevector(p0, p1, 0,1,2,3,4,5,6,7,8,9,10,11,12,13,14,15);
            p0 = *reinterpret_cast<const v8bf*>(bLo + o0);
            p1 = *reinterpret_cast<const v8bf*>(bLo + o0 + 16);
            Bl[nt] = __builtin_shufflevector(p0, p1, 0,1,2,3,4,5,6,7,8,9,10,11,12,13,14,15);
        }
#pragma unroll
        for (int mt = 0; mt < 2; ++mt)
#pragma unroll
            for (int nt = 0; nt < 4; ++nt) {
                // bf16x3 split-precision accumulation: hi*hi + hi*lo + lo*hi
                acc[mt][nt] = __builtin_amdgcn_wmma_f32_16x16x32_bf16(
                    false, Ah[mt], false, Bh[nt], (short)0, acc[mt][nt], false, false);
                acc[mt][nt] = __builtin_amdgcn_wmma_f32_16x16x32_bf16(
                    false, Ah[mt], false, Bl[nt], (short)0, acc[mt][nt], false, false);
                acc[mt][nt] = __builtin_amdgcn_wmma_f32_16x16x32_bf16(
                    false, Al[mt], false, Bh[nt], (short)0, acc[mt][nt], false, false);
            }
    };

    // prologue
    gload(0);
    stage_store(0);
    __syncthreads();

    for (int kt = 0; kt < NKT; ++kt) {
        const int cur = kt & 1;
        const bool more = (kt + 1) < NKT;
        if (more) gload(kt + 1);
        if (kt + 2 < NKT) {
            __builtin_prefetch(gA + (kt + 2) * BK, 0, 0);   // global_prefetch_b8
            __builtin_prefetch(gB + (kt + 2) * BK, 0, 0);
        }
        compute(cur);
        if (more) stage_store(cur ^ 1);
        __syncthreads();
    }

    // epilogue: C layout -> VGPR r holds M = r + 8*lh, lane%16 = N; add gathered bias
#pragma unroll
    for (int nt = 0; nt < 4; ++nt) {
        const int e = eBase + wn + nt * 16 + ln;
        const float bv = bias[(size_t)t * EMBED + e];
#pragma unroll
        for (int mt = 0; mt < 2; ++mt) {
#pragma unroll
            for (int r = 0; r < 8; ++r) {
                const int s = sBase + wm + mt * 16 + lh * 8 + r;
                out[((size_t)s * BATCH + b) * EMBED + e] = acc[mt][nt][r] + bv;
            }
        }
    }
}

extern "C" void kernel_launch(void* const* d_in, const int* in_sizes, int n_in,
                              void* d_out, int out_size, void* d_ws, size_t ws_size,
                              hipStream_t stream) {
    (void)in_sizes; (void)n_in; (void)out_size; (void)d_ws; (void)ws_size;
    const float* feat = (const float*)d_in[0];
    const float* W    = (const float*)d_in[1];
    const float* bias = (const float*)d_in[2];
    const int*   toks = (const int*)d_in[3];
    float* out = (float*)d_out;

    dim3 grid(EMBED / BN, SEQ / BM, BATCH);   // (8, 4, 128)
    grouped_linear_bf16x3_wmma<<<grid, 256, 0, stream>>>(feat, W, bias, toks, out);
}
